// MemoryEfficientAttention_62285615726758
// MI455X (gfx1250) — matrix operand order repaired and checked by reading, not compiled, not run on hardware
//
#include <hip/hip_runtime.h>
#include <hip/hip_bf16.h>
#include <math.h>
#include <stdint.h>

// ---------------------------------------------------------------------------
// MHA forward for MI455X (gfx1250, wave32): WMMA f16 16x16x32 f32-accum
// compute + Tensor Data Mover (TDM) double-buffered LDS tile staging.
// ---------------------------------------------------------------------------

typedef __attribute__((ext_vector_type(16))) _Float16 v16h;
typedef __attribute__((ext_vector_type(8)))  _Float16 v8h;
typedef __attribute__((ext_vector_type(4)))  _Float16 v4h;
typedef __attribute__((ext_vector_type(8)))  float    v8f;
typedef __attribute__((ext_vector_type(4)))  float    v4f;
typedef __attribute__((ext_vector_type(4)))  unsigned int u32x4;
typedef __attribute__((ext_vector_type(8)))  int      i32x8;
typedef __attribute__((ext_vector_type(4)))  int      i32x4;

#define HIDDEN 1024
#define SEQ    2048
#define BATCH  2
#define HEADS  16
#define HD     64
#define ROWS   (BATCH * SEQ)   // 4096

__device__ __forceinline__ v8f wmma_f16(v16h a, v16h b, v8f c) {
  return __builtin_amdgcn_wmma_f32_16x16x32_f16(
      false, a, false, b, (short)0, c, false, false);
}

// A-fragment pattern: 8 halves at p[0..7], 8 halves at p[16..23]
__device__ __forceinline__ v16h ld_frag_gap(const _Float16* p) {
  v8h lo = *(const v8h*)(p);
  v8h hi = *(const v8h*)(p + 16);
  v16h r;
#pragma unroll
  for (int i = 0; i < 8; ++i) { r[i] = lo[i]; r[i + 8] = hi[i]; }
  return r;
}

// B-fragment pattern: 8 halves at p[0..7], 8 halves at p[16..23]
__device__ __forceinline__ v16h ld_frag_seq(const _Float16* p) {
  v8h lo = *(const v8h*)(p);
  v8h hi = *(const v8h*)(p + 16);
  v16h r;
#pragma unroll
  for (int i = 0; i < 8; ++i) { r[i] = lo[i]; r[i + 8] = hi[i]; }
  return r;
}

// ------------------------- TDM 2-D tile -> LDS -----------------------------
// D# per CDNA5 ISA 08_async_tensor §8: group0 = {flags, lds_addr, gaddr,
// type=2}; group1 = {data_size=2B, tensor dims, tile dims, dim0 stride};
// groups 2/3 zero (2-D tensor). Issued by one wave; tracked by TENSORcnt.
__device__ __forceinline__ void tdm_load_2d(const void* gptr, void* lptr,
                                            unsigned tile_w, unsigned tile_h,
                                            unsigned row_stride_elems) {
  unsigned long long ga = (unsigned long long)(uintptr_t)gptr;
  unsigned lds = (unsigned)(uintptr_t)lptr;  // LDS aperture: low 32b = offset
  u32x4 g0;
  g0.x = 1u;                                         // count=1, user mode
  g0.y = lds;                                        // lds_addr (bytes)
  g0.z = (unsigned)(ga & 0xffffffffull);             // global_addr[31:0]
  g0.w = (unsigned)((ga >> 32) & 0x1ffffffull) | (2u << 30);  // [56:32]|type=2
  const unsigned td0 = row_stride_elems;             // dim0 extent (>= tile_w)
  const unsigned td1 = 1u << 20;                     // dim1 extent (large)
  i32x8 g1;
  g1[0] = (int)0x00010000u;                                   // data_size=2B
  g1[1] = (int)((td0 & 0xffffu) << 16);                       // tdim0 lo
  g1[2] = (int)(((td0 >> 16) & 0xffffu) | ((td1 & 0xffffu) << 16));
  g1[3] = (int)(((td1 >> 16) & 0xffffu) | ((tile_w & 0xffffu) << 16));
  g1[4] = (int)(tile_h & 0xffffu);                            // tile_dim1
  g1[5] = (int)row_stride_elems;                              // dim0 stride lo
  g1[6] = 0;
  g1[7] = 0;
  i32x4 z4 = {0, 0, 0, 0};
#if defined(__clang_major__) && (__clang_major__ >= 23)
  i32x8 z8 = {0, 0, 0, 0, 0, 0, 0, 0};
  __builtin_amdgcn_tensor_load_to_lds(g0, g1, z4, z4, z8, 0);
#else
  __builtin_amdgcn_tensor_load_to_lds(g0, g1, z4, z4, 0);
#endif
}

// --------------------------- fp32 -> fp16 convert ---------------------------
__global__ __launch_bounds__(256) void cvt_f32_f16_kernel(
    const float* __restrict__ src, _Float16* __restrict__ dst, int n) {
  int i = (blockIdx.x * 256 + threadIdx.x) * 4;
  if (i < n) {
    v4f a = *(const v4f*)(src + i);
    v4h h;
#pragma unroll
    for (int j = 0; j < 4; ++j) h[j] = (_Float16)a[j];
    *(v4h*)(dst + i) = h;
  }
}

// ------------------- W (k,n) f32 -> Wt (n,k) f16 transpose ------------------
__global__ __launch_bounds__(256) void transpose_w_kernel(
    const float* __restrict__ W, _Float16* __restrict__ Wt) {
  __shared__ float tile[32][33];
  int bn = blockIdx.x * 32;
  int bk = blockIdx.y * 32;
  int tx = threadIdx.x & 31;
  int ty = threadIdx.x >> 5;
#pragma unroll
  for (int i = 0; i < 4; ++i) {
    int r = ty + i * 8;
    tile[r][tx] = W[(size_t)(bk + r) * HIDDEN + bn + tx];
  }
  __syncthreads();
#pragma unroll
  for (int i = 0; i < 4; ++i) {
    int r = ty + i * 8;
    Wt[(size_t)(bn + r) * HIDDEN + bk + tx] = (_Float16)tile[tx][r];
  }
}

// ------------------------------ WMMA GEMM ----------------------------------
// C[m,n] = sum_k A[m,k] * Bt[n,k] + bias[n]; 128x64 block, 8 waves (4Mx2N),
// each wave 32x32. TDM double-buffered tiles, K-step 32.
// MODE: 0=Q(scale,BHSD) 1=K(BHSD) 2=V(BHDS) 3=final f32 [m,n]
template <int MODE>
__global__ __launch_bounds__(256) void gemm_wmma_kernel(
    const _Float16* __restrict__ A, const _Float16* __restrict__ Bt,
    const float* __restrict__ bias, void* __restrict__ out) {
  __shared__ __align__(16) _Float16 lA[2][128 * 32];
  __shared__ __align__(16) _Float16 lB[2][64 * 32];

  const int m0   = blockIdx.y * 128;
  const int n0   = blockIdx.x * 64;
  const int wave = threadIdx.x >> 5;
  const int lane = threadIdx.x & 31;
  const int half = lane >> 4;
  const int l15  = lane & 15;
  const int wm   = wave & 3;
  const int wn   = wave >> 2;

  // prologue: wave0 programs the TDM for the first K tile pair
  if (wave == 0) {
    tdm_load_2d(A + (size_t)m0 * HIDDEN, &lA[0][0], 32, 128, HIDDEN);
    tdm_load_2d(Bt + (size_t)n0 * HIDDEN, &lB[0][0], 32, 64, HIDDEN);
  }

  v8f acc[2][2] = {};
  const int NIT = HIDDEN / 32;
  for (int it = 0; it < NIT; ++it) {
    const int cur = it & 1;
    if (wave == 0) __builtin_amdgcn_s_wait_tensorcnt(0);
    __syncthreads();  // tile[cur] visible; all waves done reading tile[cur^1]
    if (wave == 0 && it + 1 < NIT) {
      const int k0n = (it + 1) * 32;
      tdm_load_2d(A + (size_t)m0 * HIDDEN + k0n, &lA[cur ^ 1][0], 32, 128,
                  HIDDEN);
      tdm_load_2d(Bt + (size_t)n0 * HIDDEN + k0n, &lB[cur ^ 1][0], 32, 64,
                  HIDDEN);
    }

    v16h af[2], bf[2];
#pragma unroll
    for (int mt = 0; mt < 2; ++mt)
      af[mt] = ld_frag_gap(&lA[cur][0] + (wm * 32 + mt * 16 + l15) * 32 +
                           half * 8);
#pragma unroll
    for (int nt = 0; nt < 2; ++nt)
      bf[nt] = ld_frag_seq(&lB[cur][0] + (wn * 32 + nt * 16 + l15) * 32 +
                           half * 8);

#pragma unroll
    for (int mt = 0; mt < 2; ++mt)
#pragma unroll
      for (int nt = 0; nt < 2; ++nt)
        acc[mt][nt] = wmma_f16(af[mt], bf[nt], acc[mt][nt]);
  }

  // epilogue: bias + layout permute
#pragma unroll
  for (int mt = 0; mt < 2; ++mt) {
#pragma unroll
    for (int nt = 0; nt < 2; ++nt) {
#pragma unroll
      for (int r = 0; r < 8; ++r) {
        int m = m0 + wm * 32 + mt * 16 + r + half * 8;
        int n = n0 + wn * 32 + nt * 16 + l15;
        float v = acc[mt][nt][r] + bias[n];
        if constexpr (MODE == 0 || MODE == 1) {
          if constexpr (MODE == 0) v *= 0.125f;  // fold 1/sqrt(HD) into Q
          int b = m >> 11, s = m & 2047, h = n >> 6, d = n & 63;
          ((_Float16*)out)[(((size_t)(b * HEADS + h)) * SEQ + s) * HD + d] =
              (_Float16)v;
        } else if constexpr (MODE == 2) {
          int b = m >> 11, s = m & 2047, h = n >> 6, d = n & 63;
          ((_Float16*)out)[(((size_t)(b * HEADS + h)) * HD + d) * SEQ + s] =
              (_Float16)v;
        } else {
          ((float*)out)[(size_t)m * HIDDEN + n] = v;
        }
      }
    }
  }
}

// ---------------------------- Flash attention ------------------------------
// grid = (SEQ/128, BATCH*HEADS); 8 waves x 16 query rows x full d=64.
// Q pre-scaled by 1/8; K = [b,h,s,d]; V pre-transposed [b,h,d,s].
// K/V blocks staged by TDM with double buffering.
__global__ __launch_bounds__(256) void attn_kernel(
    const _Float16* __restrict__ qh, const _Float16* __restrict__ kh,
    const _Float16* __restrict__ vth, _Float16* __restrict__ ctxh) {
  __shared__ __align__(16) _Float16 ldsK[2][32 * 64];   // [key][d]
  __shared__ __align__(16) _Float16 ldsV[2][64 * 32];   // [d][key]
  __shared__ __align__(16) _Float16 ldsP[8 * 16 * 40];  // per-wave P transpose

  const int bh   = blockIdx.y;
  const int q0   = blockIdx.x * 128;
  const int wave = threadIdx.x >> 5;
  const int lane = threadIdx.x & 31;
  const int half = lane >> 4;
  const int l15  = lane & 15;
  const int qrow = q0 + wave * 16;

  const _Float16* kbase = kh + (size_t)bh * SEQ * HD;
  const _Float16* vbase = vth + (size_t)bh * HD * SEQ;

  // Q A-fragments straight from global (d contiguous per lane)
  const _Float16* qbase = qh + ((size_t)bh * SEQ + qrow) * HD;
  v16h qf[2];
#pragma unroll
  for (int f = 0; f < 2; ++f)
    qf[f] = ld_frag_gap(qbase + (size_t)l15 * HD + f * 32 + half * 8);

  v8f O[4] = {};
  float m8[8], l8[8];
#pragma unroll
  for (int r = 0; r < 8; ++r) { m8[r] = -__builtin_inff(); l8[r] = 0.0f; }

  // prologue TDM: first 32-key block
  if (wave == 0) {
    tdm_load_2d(kbase, &ldsK[0][0], 64, 32, 64);        // 32 keys x 64 d
    tdm_load_2d(vbase, &ldsV[0][0], 32, 64, SEQ);       // 64 d x 32 keys
  }

  const int NIT = SEQ / 32;
  for (int it = 0; it < NIT; ++it) {
    const int cur = it & 1;
    if (wave == 0) __builtin_amdgcn_s_wait_tensorcnt(0);
    __syncthreads();
    if (wave == 0 && it + 1 < NIT) {
      const int nb = (it + 1) * 32;
      tdm_load_2d(kbase + (size_t)nb * HD, &ldsK[cur ^ 1][0], 64, 32, 64);
      tdm_load_2d(vbase + nb, &ldsV[cur ^ 1][0], 32, 64, SEQ);
    }
    const _Float16* K = &ldsK[cur][0];
    const _Float16* V = &ldsV[cur][0];

    // S = Q K^T (two 16-key N tiles; d=64 split into two x32 WMMAs)
    v8f S0 = {}, S1 = {};
    {
      v16h k00 = ld_frag_seq(K + (0 + l15) * 64 + 0 + half * 8);
      v16h k01 = ld_frag_seq(K + (0 + l15) * 64 + 32 + half * 8);
      S0 = wmma_f16(qf[0], k00, S0);
      S0 = wmma_f16(qf[1], k01, S0);
      v16h k10 = ld_frag_seq(K + (16 + l15) * 64 + 0 + half * 8);
      v16h k11 = ld_frag_seq(K + (16 + l15) * 64 + 32 + half * 8);
      S1 = wmma_f16(qf[0], k10, S1);
      S1 = wmma_f16(qf[1], k11, S1);
    }

    // online softmax row stats (row spans 16 lanes in C layout)
    float mx[8], mnew[8], corr[8], rs[8], p0[8], p1[8];
#pragma unroll
    for (int r = 0; r < 8; ++r) mx[r] = fmaxf(S0[r], S1[r]);
#pragma unroll
    for (int r = 0; r < 8; ++r) {
#pragma unroll
      for (int mask = 1; mask < 16; mask <<= 1)
        mx[r] = fmaxf(mx[r], __shfl_xor(mx[r], mask, 32));
    }
#pragma unroll
    for (int r = 0; r < 8; ++r) {
      mnew[r] = fmaxf(m8[r], mx[r]);
      corr[r] = __expf(m8[r] - mnew[r]);
      p0[r]   = __expf(S0[r] - mnew[r]);
      p1[r]   = __expf(S1[r] - mnew[r]);
      rs[r]   = p0[r] + p1[r];
    }
#pragma unroll
    for (int r = 0; r < 8; ++r) {
#pragma unroll
      for (int mask = 1; mask < 16; mask <<= 1)
        rs[r] += __shfl_xor(rs[r], mask, 32);
    }
#pragma unroll
    for (int r = 0; r < 8; ++r) {
      l8[r] = l8[r] * corr[r] + rs[r];
      m8[r] = mnew[r];
    }
#pragma unroll
    for (int d = 0; d < 4; ++d)
#pragma unroll
      for (int r = 0; r < 8; ++r) O[d][r] *= corr[r];

    // P: C layout -> per-wave LDS -> A fragment
    _Float16* pw = ldsP + wave * (16 * 40);
#pragma unroll
    for (int r = 0; r < 8; ++r) {
      int M = r + half * 8;
      pw[M * 40 + l15]      = (_Float16)p0[r];
      pw[M * 40 + 16 + l15] = (_Float16)p1[r];
    }
    v16h pf = ld_frag_gap(pw + l15 * 40 + half * 8);

    // O += P V  (K-dim = 32 keys, 4 d-tiles)
#pragma unroll
    for (int d = 0; d < 4; ++d) {
      v16h vf = ld_frag_seq(V + (d * 16 + l15) * 32 + half * 8);
      O[d] = wmma_f16(pf, vf, O[d]);
    }
  }

  // epilogue: normalize + store ctx [b, s, h*64+d] f16
  const int b = bh >> 4, h = bh & 15;
  float rinv[8];
#pragma unroll
  for (int r = 0; r < 8; ++r) rinv[r] = 1.0f / l8[r];
#pragma unroll
  for (int d = 0; d < 4; ++d) {
#pragma unroll
    for (int r = 0; r < 8; ++r) {
      int q = qrow + r + half * 8;
      int c = d * 16 + l15;
      ctxh[((size_t)(b * SEQ + q)) * HIDDEN + h * HD + c] =
          (_Float16)(O[d][r] * rinv[r]);
    }
  }
}

// ------------------------------- launcher ----------------------------------
extern "C" void kernel_launch(void* const* d_in, const int* in_sizes, int n_in,
                              void* d_out, int out_size, void* d_ws,
                              size_t ws_size, hipStream_t stream) {
  const float* x  = (const float*)d_in[0];
  const float* Wq = (const float*)d_in[1];
  const float* bq = (const float*)d_in[2];
  const float* Wk = (const float*)d_in[3];
  const float* bk = (const float*)d_in[4];
  const float* Wv = (const float*)d_in[5];
  const float* bv = (const float*)d_in[6];
  const float* Wo = (const float*)d_in[7];
  const float* bo = (const float*)d_in[8];

  char* w = (char*)d_ws;
  const size_t SZ_ACT = (size_t)ROWS * HIDDEN * sizeof(_Float16);
  const size_t SZ_W   = (size_t)HIDDEN * HIDDEN * sizeof(_Float16);
  _Float16* xh   = (_Float16*)w; w += SZ_ACT;
  _Float16* wqt  = (_Float16*)w; w += SZ_W;
  _Float16* wkt  = (_Float16*)w; w += SZ_W;
  _Float16* wvt  = (_Float16*)w; w += SZ_W;
  _Float16* wot  = (_Float16*)w; w += SZ_W;
  _Float16* q_h  = (_Float16*)w; w += SZ_ACT;  // [b,h,s,d], pre-scaled
  _Float16* k_h  = (_Float16*)w; w += SZ_ACT;  // [b,h,s,d]
  _Float16* vt_h = (_Float16*)w; w += SZ_ACT;  // [b,h,d,s]
  _Float16* ctx  = (_Float16*)w; w += SZ_ACT;  // [b*s, h*d]

  {
    int n = ROWS * HIDDEN;
    cvt_f32_f16_kernel<<<n / (256 * 4), 256, 0, stream>>>(x, xh, n);
    dim3 g(32, 32);
    transpose_w_kernel<<<g, 256, 0, stream>>>(Wq, wqt);
    transpose_w_kernel<<<g, 256, 0, stream>>>(Wk, wkt);
    transpose_w_kernel<<<g, 256, 0, stream>>>(Wv, wvt);
    transpose_w_kernel<<<g, 256, 0, stream>>>(Wo, wot);
  }
  {
    dim3 g(HIDDEN / 64, ROWS / 128);
    gemm_wmma_kernel<0><<<g, 256, 0, stream>>>(xh, wqt, bq, (void*)q_h);
    gemm_wmma_kernel<1><<<g, 256, 0, stream>>>(xh, wkt, bk, (void*)k_h);
    gemm_wmma_kernel<2><<<g, 256, 0, stream>>>(xh, wvt, bv, (void*)vt_h);
  }
  {
    dim3 g(SEQ / 128, BATCH * HEADS);
    attn_kernel<<<g, 256, 0, stream>>>(q_h, k_h, vt_h, ctx);
  }
  {
    dim3 g(HIDDEN / 64, ROWS / 128);
    gemm_wmma_kernel<3><<<g, 256, 0, stream>>>(ctx, wot, bo, d_out);
  }
}
